// TrackTokenGater_88596585382096
// MI455X (gfx1250) — matrix operand order repaired and checked by reading, not compiled, not run on hardware
//
#include <hip/hip_runtime.h>

// ---------------- problem constants ----------------
#define NTOK   64
#define DDIM   1024
#define HDIM   512
#define TOPKN  15
#define KCHUNK 256                 // K elems staged per chunk
#define NCHUNKS (DDIM / KCHUNK)    // 4
#define KSTEPS  (KCHUNK / 32)      // 8 wmma K-steps per chunk
#define NTILES  (HDIM / 16)        // 32
#define MTILES  (NTOK / 16)        // 4
#define BLOCK   256                // 8 waves (wave32)

// padded LDS strides (bank-conflict avoidance)
#define TOK_STRIDE 1028            // floats per token row (1024 + 4 pad)
#define XN_STRIDE  264             // bf16 per token row   (256 + 8 pad)

// LDS layout (bytes)
#define OFF_TOK    0
#define OFF_XN     (NTOK * TOK_STRIDE * 4)                 // 263168
#define OFF_GAMMA  (OFF_XN + NTOK * XN_STRIDE * 2)         // 296960
#define OFF_BETA   (OFF_GAMMA + DDIM * 4)                  // 301056
#define OFF_MU     (OFF_BETA + DDIM * 4)                   // 305152
#define OFF_RSTD   (OFF_MU + NTOK * 4)                     // 305408
#define OFF_SCORES (OFF_RSTD + NTOK * 4)                   // 305664
#define OFF_TOPK   (OFF_SCORES + NTOK * 4)                 // 305920
#define OFF_SEL    (OFF_TOPK + 16 * 4)                     // 305984
#define OFF_BGC    (OFF_SEL + NTOK * 4)                    // 306240
#define SMEM_TOTAL (OFF_BGC + 16)                          // 306256 < 320KB

typedef __attribute__((ext_vector_type(16))) __bf16 v16bf;
typedef __attribute__((ext_vector_type(8)))  float  v8f;
typedef int nint4 __attribute__((vector_size(16)));   // clang-native int4

union BF16x16 { uint4 q[2]; unsigned short us[16]; v16bf v; };
union PackBF  { unsigned short h[2]; unsigned u; };

#if __has_builtin(__builtin_amdgcn_global_load_async_to_lds_b128)
#define HAVE_ASYNC_LDS 1
// builtin takes native-vector int4* for both global src and LDS dst
#define ASYNC_CP_B128(gsrc, ldst) \
  __builtin_amdgcn_global_load_async_to_lds_b128((nint4*)(gsrc), (nint4*)(ldst), 0, 0)
#endif

__device__ __forceinline__ void wait_asynccnt0() {
#if __has_builtin(__builtin_amdgcn_s_wait_asynccnt)
  __builtin_amdgcn_s_wait_asynccnt(0);
#else
  asm volatile("s_wait_asynccnt 0x0" ::: "memory");
#endif
}

__device__ __forceinline__ unsigned short f2bf(float f) {
  unsigned u = __float_as_uint(f);
  u += 0x7FFFu + ((u >> 16) & 1u);   // round-to-nearest-even
  return (unsigned short)(u >> 16);
}

// ---------------------------------------------------------------------------
// Prep: convert w1 [D x H] fp32 -> bf16 in WMMA B-fragment tile order.
// Tile (kt in 0..31, nt in 0..31); within tile: lane l (0..31) holds 16
// contiguous bf16: n = nt*16 + (l&15); k = kt*32 + (e<8 ? 8*(l>>4)+e : 16+8*(l>>4)+(e-8)).
// ---------------------------------------------------------------------------
__global__ void prep_w1_kernel(const float* __restrict__ w1,
                               unsigned short* __restrict__ w1s) {
  int o = blockIdx.x * blockDim.x + threadIdx.x;
  if (o >= DDIM * HDIM) return;
  int e  = o & 15;
  int l  = (o >> 4) & 31;
  int nt = (o >> 9) & 31;
  int kt = o >> 14;
  int n = nt * 16 + (l & 15);
  int k = kt * 32 + ((e & 8) << 1) + ((l >> 4) << 3) + (e & 7);
  w1s[o] = f2bf(w1[k * HDIM + n]);
}

// ---------------------------------------------------------------------------
// Main kernel: one workgroup per batch.
// ---------------------------------------------------------------------------
__global__ void __launch_bounds__(BLOCK)
gater_kernel(const float* __restrict__ tokens,
             const float* __restrict__ gamma,
             const float* __restrict__ beta,
             const unsigned short* __restrict__ w1s,
             const float* __restrict__ b1,
             const float* __restrict__ w2,
             float* __restrict__ out) {
  extern __shared__ char smem[];
  float*          tokL   = (float*)(smem + OFF_TOK);
  unsigned short* xnL    = (unsigned short*)(smem + OFF_XN);
  float*          gammaL = (float*)(smem + OFF_GAMMA);
  float*          betaL  = (float*)(smem + OFF_BETA);
  float*          muL    = (float*)(smem + OFF_MU);
  float*          rstdL  = (float*)(smem + OFF_RSTD);
  float*          scoresL= (float*)(smem + OFF_SCORES);
  int*            topkL  = (int*)(smem + OFF_TOPK);
  int*            selL   = (int*)(smem + OFF_SEL);
  float*          bgcL   = (float*)(smem + OFF_BGC);

  const int t    = threadIdx.x;
  const int b    = blockIdx.x;
  const int lane = t & 31;
  const int wid  = t >> 5;

  // ---- 1. stream the 64x1024 fp32 token tile into LDS (read HBM once) ----
#ifdef HAVE_ASYNC_LDS
  {
    // CDNA5 direct-to-LDS async copy: no VGPR round trip, tracked by ASYNCcnt.
    const float4* src = (const float4*)(tokens + (size_t)b * NTOK * DDIM);
    #pragma unroll 8
    for (int i = 0; i < (NTOK * DDIM / 4) / BLOCK; ++i) {     // 64 iters
      int g4 = t + BLOCK * i;                                  // float4 index
      int n  = g4 >> 8;                                        // token
      int c4 = g4 & 255;                                       // float4 col
      ASYNC_CP_B128(src + g4, tokL + n * TOK_STRIDE + c4 * 4);
    }
    // gamma/beta: 1024 floats each
    const float4* g4p = (const float4*)gamma;
    const float4* b4p = (const float4*)beta;
    for (int i = t; i < DDIM / 4; i += BLOCK) {
      ASYNC_CP_B128(g4p + i, gammaL + i * 4);
      ASYNC_CP_B128(b4p + i, betaL + i * 4);
    }
    wait_asynccnt0();
  }
#else
  {
    const float4* src = (const float4*)(tokens + (size_t)b * NTOK * DDIM);
    for (int i = 0; i < (NTOK * DDIM / 4) / BLOCK; ++i) {     // 64 iters
      int g4  = t + BLOCK * i;
      int n   = g4 >> 8;
      int c4  = g4 & 255;
      *(float4*)(tokL + n * TOK_STRIDE + c4 * 4) = src[g4];
    }
    for (int i = t; i < DDIM; i += BLOCK) {
      gammaL[i] = gamma[i];
      betaL[i]  = beta[i];
    }
  }
#endif
  __syncthreads();

  // ---- 2. LayerNorm stats: 4 threads per token ----
  {
    int n = t >> 2, q = t & 3;
    const float4* row = (const float4*)(tokL + n * TOK_STRIDE + q * 256);
    float s = 0.f, s2 = 0.f;
    #pragma unroll 8
    for (int i = 0; i < 64; ++i) {
      float4 v = row[i];
      s  += v.x + v.y + v.z + v.w;
      s2 += v.x * v.x + v.y * v.y + v.z * v.z + v.w * v.w;
    }
    s  += __shfl_xor(s, 1, 32);  s  += __shfl_xor(s, 2, 32);
    s2 += __shfl_xor(s2, 1, 32); s2 += __shfl_xor(s2, 2, 32);
    if (q == 0) {
      float mu  = s * (1.f / DDIM);
      float var = s2 * (1.f / DDIM) - mu * mu;
      muL[n]   = mu;
      rstdL[n] = rsqrtf(var + 1e-5f);
    }
    if (t < NTOK) scoresL[t] = 0.f;
  }
  __syncthreads();

  // ---- 3. Scoring GEMM: h = relu(xn@w1+b1); scores = h@w2  (WMMA bf16) ----
  const int ntBase = wid * 4;            // each wave owns 4 N-tiles
  v8f acc[MTILES][4];
  #pragma unroll
  for (int mt = 0; mt < MTILES; ++mt)
    #pragma unroll
    for (int j = 0; j < 4; ++j)
      #pragma unroll
      for (int r = 0; r < 8; ++r) acc[mt][j][r] = 0.f;

  for (int ch = 0; ch < NCHUNKS; ++ch) {
    // 3a. cooperatively stage normalized bf16 chunk (fused LayerNorm)
    {
      int n  = t >> 2;
      int kb = (t & 3) * 64;
      float mu = muL[n], rs = rstdL[n];
      const float* trow = tokL + n * TOK_STRIDE + ch * KCHUNK;
      unsigned* xrow = (unsigned*)(xnL + n * XN_STRIDE);   // XN_STRIDE even
      #pragma unroll 8
      for (int i = 0; i < 32; ++i) {
        int kk = kb + 2 * i;
        int kg = ch * KCHUNK + kk;
        float x0 = (trow[kk]     - mu) * rs * gammaL[kg]     + betaL[kg];
        float x1 = (trow[kk + 1] - mu) * rs * gammaL[kg + 1] + betaL[kg + 1];
        PackBF p; p.h[0] = f2bf(x0); p.h[1] = f2bf(x1);
        xrow[kk >> 1] = p.u;
      }
    }
    __syncthreads();

    // 3b. WMMA over the 8 K-steps of this chunk
    for (int ks = 0; ks < KSTEPS; ++ks) {
      int ktG = ch * KSTEPS + ks;
      int kb  = ks * 32;
      int o1  = kb + ((lane < 16) ? 0 : 8);    // A-frag K offsets per ISA
      int o2  = kb + ((lane < 16) ? 16 : 24);
      int mrow = lane & 15;
      BF16x16 A[MTILES];
      #pragma unroll
      for (int mt = 0; mt < MTILES; ++mt) {
        const unsigned short* rp = xnL + (mt * 16 + mrow) * XN_STRIDE;
        A[mt].q[0] = *(const uint4*)(rp + o1);
        A[mt].q[1] = *(const uint4*)(rp + o2);
      }
      #pragma unroll
      for (int j = 0; j < 4; ++j) {
        BF16x16 Bf;
        const uint4* bp = (const uint4*)
            (w1s + ((((size_t)ktG * NTILES) + (ntBase + j)) * 32 + lane) * 16);
        Bf.q[0] = bp[0];
        Bf.q[1] = bp[1];
        #pragma unroll
        for (int mt = 0; mt < MTILES; ++mt)
          acc[mt][j] = __builtin_amdgcn_wmma_f32_16x16x32_bf16(
              false, A[mt].v, false, Bf.v, (short)0, acc[mt][j], false, false);
      }
    }
    __syncthreads();   // waves done reading xn before next chunk overwrites
  }

  // ---- 4. bias + ReLU + dot(w2), reduce to per-token scores ----
  {
    float contrib[MTILES][8];
    #pragma unroll
    for (int mt = 0; mt < MTILES; ++mt)
      #pragma unroll
      for (int r = 0; r < 8; ++r) contrib[mt][r] = 0.f;

    #pragma unroll
    for (int j = 0; j < 4; ++j) {
      int h = (ntBase + j) * 16 + (lane & 15);   // hidden index (C: N = lane%16)
      float b1v = b1[h], w2v = w2[h];
      #pragma unroll
      for (int mt = 0; mt < MTILES; ++mt)
        #pragma unroll
        for (int r = 0; r < 8; ++r) {
          float v = acc[mt][j][r] + b1v;
          v = v > 0.f ? v : 0.f;
          contrib[mt][r] += v * w2v;
        }
    }
    // butterfly across the 16 lanes of each half (C: M = mt*16 + r + 8*(lane>=16))
    #pragma unroll
    for (int mt = 0; mt < MTILES; ++mt)
      #pragma unroll
      for (int r = 0; r < 8; ++r) {
        float v = contrib[mt][r];
        v += __shfl_xor(v, 1, 32);
        v += __shfl_xor(v, 2, 32);
        v += __shfl_xor(v, 4, 32);
        v += __shfl_xor(v, 8, 32);
        if ((lane & 15) == 0)
          atomicAdd(&scoresL[mt * 16 + r + ((lane >> 4) << 3)], v);
      }
  }
  __syncthreads();

  // ---- 5. serial top-15 (strict '>' == JAX tie-break: lowest index wins) ----
  if (t == 0) {
    for (int j = 0; j < NTOK; ++j) selL[j] = 0;
    for (int i = 0; i < TOPKN; ++i) {
      float best = -3.4e38f; int bi = 0;
      for (int j = 0; j < NTOK; ++j) {
        float v = scoresL[j];
        if (v > best) { best = v; bi = j; }
      }
      topkL[i] = bi;
      scoresL[bi] = -3.4e38f;
      selL[bi] = 1;
    }
    float c = (float)(NTOK - TOPKN);
    bgcL[0] = (c < 1.f) ? 1.f : c;
  }
  __syncthreads();

  // ---- 6. outputs: row 0 = background mean, rows 1..15 = gathered top-k ----
  float* ob = out + (size_t)b * (TOPKN + 1) * DDIM;
  float invc = 1.f / bgcL[0];
  #pragma unroll
  for (int i = 0; i < DDIM / BLOCK; ++i) {     // 4 cols per thread
    int col = t + BLOCK * i;
    float s = 0.f;
    for (int n = 0; n < NTOK; ++n)
      if (!selL[n]) s += tokL[n * TOK_STRIDE + col];
    ob[col] = s * invc;
  }
  for (int e = t; e < TOPKN * DDIM; e += BLOCK) {
    int row = e >> 10, col = e & (DDIM - 1);
    ob[(row + 1) * DDIM + col] = tokL[topkL[row] * TOK_STRIDE + col];
  }
}

// ---------------------------------------------------------------------------
extern "C" void kernel_launch(void* const* d_in, const int* in_sizes, int n_in,
                              void* d_out, int out_size, void* d_ws, size_t ws_size,
                              hipStream_t stream) {
  const float* tokens = (const float*)d_in[0];
  const float* gamma  = (const float*)d_in[1];
  const float* beta   = (const float*)d_in[2];
  const float* w1     = (const float*)d_in[3];
  const float* b1     = (const float*)d_in[4];
  const float* w2     = (const float*)d_in[5];
  // d_in[6] = b2: constant shift of all scores -> cannot change top-k indices
  // or any output value; intentionally unused.

  int B = in_sizes[0] / (NTOK * DDIM);
  unsigned short* w1s = (unsigned short*)d_ws;   // 1024*512*2 = 1 MB swizzled bf16 w1

  prep_w1_kernel<<<(DDIM * HDIM + 255) / 256, 256, 0, stream>>>(w1, w1s);
  gater_kernel<<<B, BLOCK, SMEM_TOTAL, stream>>>(tokens, gamma, beta, w1s,
                                                 b1, w2, (float*)d_out);
}
